// SSMWorld_37014028157371
// MI455X (gfx1250) — matrix operand order repaired and checked by reading, not compiled, not run on hardware
//
#include <hip/hip_runtime.h>
#include <hip/hip_bf16.h>

typedef __attribute__((ext_vector_type(16))) __bf16       v16bf;
typedef __attribute__((ext_vector_type(8)))  float        v8f;
typedef __attribute__((ext_vector_type(8)))  unsigned int v8u;
typedef __attribute__((ext_vector_type(4)))  unsigned int v4u;
typedef __attribute__((ext_vector_type(4)))  int          v4i;

struct BF16Tile { unsigned short h[16]; };

#ifndef __has_builtin
#define __has_builtin(x) 0
#endif

#if __has_builtin(__builtin_amdgcn_global_load_async_to_lds_b128)
#define HAVE_ASYNC_B128 1
#else
#define HAVE_ASYNC_B128 0
#endif

typedef __attribute__((address_space(1))) v4i g_v4i;   // global (AS1) int4
typedef __attribute__((address_space(3))) v4i l_v4i;   // LDS (AS3) int4

// 16B global -> LDS copy; async (ASYNCcnt) when available.
__device__ __forceinline__ void copy16_g2l(void* lds_dst, const void* g_src) {
#if HAVE_ASYNC_B128
  void* gs = const_cast<void*>(g_src);
  __builtin_amdgcn_global_load_async_to_lds_b128((g_v4i*)gs, (l_v4i*)lds_dst, 0, 0);
#else
  *(v4u*)lds_dst = *(const v4u*)g_src;
#endif
}

__device__ __forceinline__ void async_wait0() {
#if __has_builtin(__builtin_amdgcn_s_wait_asynccnt)
  __builtin_amdgcn_s_wait_asynccnt(0);
#elif HAVE_ASYNC_B128
  asm volatile("s_wait_asynccnt 0x0" ::: "memory");
#endif
}

__device__ __forceinline__ unsigned short f2bf(float f) {
  unsigned int u = __float_as_uint(f);
  unsigned int r = u + 0x7FFFu + ((u >> 16) & 1u);   // round-to-nearest-even
  return (unsigned short)(r >> 16);
}

// ---- WMMA operand loaders (16x16x32 bf16; contiguous k = 16*(lane/16) + s in both A and B) ----

// A-matrix 16x32 bf16 from a bf16 row-major source (16 rows x ldk), tile K offset k0.
// lane 0-15: M=lane, K = c0..c0+15, c0=0 ; lane 16-31: c0=16.
__device__ __forceinline__ v16bf load_a_bf16(const unsigned short* base, int ldk, int k0, int lane) {
  const int m  = lane & 15;
  const int c0 = (lane < 16) ? 0 : 16;
  const unsigned short* row = base + (size_t)m * ldk + k0 + c0;
  v4u lo = *(const v4u*)(row);        // 8 bf16
  v4u hi = *(const v4u*)(row + 8);    // 8 bf16
  BF16Tile t;
  *(v4u*)(&t.h[0]) = lo;
  *(v4u*)(&t.h[8]) = hi;
  return __builtin_bit_cast(v16bf, t);
}

// B-matrix 32x16 bf16; source stored [col][k] row-major.
// lane 0-15: N=lane, K=k0..k0+15 ; lane 16-31: N=lane-16, K=k0+16..k0+31 (16 contiguous bf16).
__device__ __forceinline__ v16bf load_b_bf16(const unsigned short* base, int ldk, int k0,
                                             int colBase, int lane) {
  const int col  = colBase + (lane & 15);
  const int koff = k0 + ((lane < 16) ? 0 : 16);
  v8u raw = *(const v8u*)(base + (size_t)col * ldk + koff);
  return __builtin_bit_cast(v16bf, raw);
}

__device__ __forceinline__ v8f wmma_bf16(v16bf a, v16bf b, v8f c) {
  return __builtin_amdgcn_wmma_f32_16x16x32_bf16(false, a, false, b, (short)0, c, false, false);
}

// =================== kernels ===================

__global__ void k_f32_to_bf16(const float* __restrict__ in, unsigned short* __restrict__ out, int n) {
  int i = blockIdx.x * blockDim.x + threadIdx.x;
  if (i < n) out[i] = f2bf(in[i]);
}

// Vectorized x conversion: 8 fp32 -> 8 bf16 per thread.
__global__ void k_x_to_bf16(const float* __restrict__ in, unsigned short* __restrict__ out, int n8) {
  int i = blockIdx.x * blockDim.x + threadIdx.x;
  if (i >= n8) return;
  v8f v = *(const v8f*)(in + (size_t)i * 8);
  BF16Tile t;
#pragma unroll
  for (int j = 0; j < 8; ++j) t.h[j] = f2bf(v[j]);
  *(v4u*)(out + (size_t)i * 8) = *(v4u*)t.h;
}

// bx[s,b,n] = sum_d x[b,s,d] * B_w[n,d].  xbf [B*S,512] bf16; Bw [256,512] bf16; bx [S,B,N] f32.
// grid: 2048 blocks (32 tokens = 2 m-tiles), 256 threads = 8 waves; wave w -> n-tiles 2w,2w+1.
__global__ void k_bx_gemm(const unsigned short* __restrict__ xbf, const unsigned short* __restrict__ Bw,
                          float* __restrict__ bx) {
  const int lane = threadIdx.x & 31;
  const int wave = threadIdx.x >> 5;
  const int t0   = blockIdx.x * 32;                 // token = b*1024 + s
  const unsigned short* arow0 = xbf + (size_t)t0 * 512;
  const unsigned short* arow1 = arow0 + 16 * 512;
  const int n0 = wave * 32;
  v8f acc00 = {}, acc01 = {}, acc10 = {}, acc11 = {};
#pragma unroll 4
  for (int k0 = 0; k0 < 512; k0 += 32) {
    v16bf a0 = load_a_bf16(arow0, 512, k0, lane);
    v16bf a1 = load_a_bf16(arow1, 512, k0, lane);
    v16bf b0 = load_b_bf16(Bw, 512, k0, n0,      lane);
    v16bf b1 = load_b_bf16(Bw, 512, k0, n0 + 16, lane);
    acc00 = wmma_bf16(a0, b0, acc00);
    acc01 = wmma_bf16(a0, b1, acc01);
    acc10 = wmma_bf16(a1, b0, acc10);
    acc11 = wmma_bf16(a1, b1, acc11);
  }
  const int hi = (lane < 16) ? 0 : 8;
  const int c  = lane & 15;
#pragma unroll
  for (int r = 0; r < 8; ++r) {
    int t = t0 + r + hi;
    int b = t >> 10, s = t & 1023;
    float* dst = bx + ((size_t)((s << 6) + b) << 8);  // (s*64+b)*256
    dst[n0 + c]      = acc00[r];
    dst[n0 + 16 + c] = acc01[r];
    t += 16; b = t >> 10; s = t & 1023;
    dst = bx + ((size_t)((s << 6) + b) << 8);
    dst[n0 + c]      = acc10[r];
    dst[n0 + 16 + c] = acc11[r];
  }
}

// Serial scan: state = clip(state @ A^T + bx_s).  4 blocks x 16 batch rows.
// A bf16 resident in LDS (padded rows); bf16 state ping-pong in LDS; bx slice double-buffered
// in LDS via async global->LDS loads overlapped with the WMMAs of the current step.
#define ALD 272   // A row stride in bf16 (544 B: 32B-aligned, bank-shifted)
#define STLD 272  // state row stride in bf16
__global__ void k_scan(const unsigned short* __restrict__ Abf, const float* __restrict__ bx,
                       unsigned short* __restrict__ st_out) {
  extern __shared__ unsigned char smem[];
  unsigned short* Alds = (unsigned short*)smem;                          // 256*ALD bf16 = 139264 B
  unsigned short* st0  = (unsigned short*)(smem + 256 * ALD * 2);        // 16*STLD bf16
  unsigned short* st1  = st0 + 16 * STLD;
  float* bxb0 = (float*)(st1 + 16 * STLD);                               // 16*256 f32 = 16 KB
  float* bxb1 = bxb0 + 16 * 256;

  const int lane = threadIdx.x & 31;
  const int wave = threadIdx.x >> 5;
  const int bg   = blockIdx.x * 16;

  // Stage A (128 KB packed -> padded rows) through the async path.
  for (int ci = threadIdx.x; ci < 8192; ci += blockDim.x) {      // 8192 x 16B
    int r = ci >> 5, cc = ci & 31;
    copy16_g2l((unsigned char*)Alds + r * (ALD * 2) + cc * 16,
               (const unsigned char*)Abf + (size_t)ci * 16);
  }
  // Preload bx slice for s=0.
  for (int ci = threadIdx.x; ci < 1024; ci += blockDim.x)        // 16 KB
    copy16_g2l((unsigned char*)bxb0 + ci * 16,
               (const unsigned char*)(bx + ((size_t)bg << 8)) + ci * 16);
  // Zero initial state (both buffers).
  for (int i = threadIdx.x; i < 16 * STLD; i += blockDim.x) { st0[i] = 0; st1[i] = 0; }

  const int n0 = wave * 32;
  const int hi = (lane < 16) ? 0 : 8;
  const int c  = lane & 15;
  unsigned short* cur = st0; unsigned short* nxt = st1;
  float* bxcur = bxb0; float* bxnxt = bxb1;

  for (int s = 0; s < 1024; ++s) {
    async_wait0();       // this wave's async copies into the current bx buffer done
    __syncthreads();     // all waves' copies + previous step's state writes visible

    if (s + 1 < 1024) {  // kick async load of the next bx slice into the other buffer
      const float* src = bx + ((size_t)(((s + 1) << 6) + bg) << 8);
      for (int ci = threadIdx.x; ci < 1024; ci += blockDim.x)
        copy16_g2l((unsigned char*)bxnxt + ci * 16, (const unsigned char*)src + ci * 16);
    }

    v8f acc0 = {}; v8f acc1 = {};
#pragma unroll
    for (int k0 = 0; k0 < 256; k0 += 32) {
      v16bf a  = load_a_bf16(cur, STLD, k0, lane);            // bf16 state from LDS
      v16bf b0 = load_b_bf16(Alds, ALD, k0, n0,      lane);   // B[k][n] = A[n][k]
      v16bf b1 = load_b_bf16(Alds, ALD, k0, n0 + 16, lane);
      acc0 = wmma_bf16(a, b0, acc0);
      acc1 = wmma_bf16(a, b1, acc1);
    }
#pragma unroll
    for (int r = 0; r < 8; ++r) {
      int m = r + hi;
      float v0 = acc0[r] + bxcur[m * 256 + n0 + c];
      float v1 = acc1[r] + bxcur[m * 256 + n0 + 16 + c];
      v0 = fminf(fmaxf(v0, -10.f), 10.f);
      v1 = fminf(fmaxf(v1, -10.f), 10.f);
      unsigned short h0 = f2bf(v0), h1 = f2bf(v1);
      nxt[m * STLD + n0 + c]      = h0;
      nxt[m * STLD + n0 + 16 + c] = h1;
      size_t go = ((((size_t)(bg + m)) << 10) | (unsigned)s) << 8;  // (b*1024+s)*256
      st_out[go + n0 + c]      = h0;
      st_out[go + n0 + 16 + c] = h1;
    }
    unsigned short* t1 = cur; cur = nxt; nxt = t1;
    float* t2 = bxcur; bxcur = bxnxt; bxnxt = t2;
  }
}

// Fused: y = st@C_w^T + x@D_w^T + x, LayerNorm over D=512, fp32 out.
// grid: 2048 blocks x 32 tokens (2 m-tiles); wave w -> d-cols [64w, 64w+64) (4 tiles).
// Residual is folded into the accumulators; LN makes two passes over live accumulators.
__global__ void k_out_ln(const unsigned short* __restrict__ stbf, const unsigned short* __restrict__ Cw,
                         const unsigned short* __restrict__ xbf, const float* __restrict__ x,
                         const unsigned short* __restrict__ Dw,
                         const float* __restrict__ gamma, const float* __restrict__ beta,
                         float* __restrict__ out) {
  __shared__ float s_sum[32];
  __shared__ float s_sq[32];
  const int lane = threadIdx.x & 31;
  const int wave = threadIdx.x >> 5;
  const int t0   = blockIdx.x * 32;
  const int d0   = wave * 64;
  v8f acc[2][4] = {};

  const unsigned short* arow_s0 = stbf + (size_t)t0 * 256;
  const unsigned short* arow_s1 = arow_s0 + 16 * 256;
#pragma unroll
  for (int k0 = 0; k0 < 256; k0 += 32) {
    v16bf a0 = load_a_bf16(arow_s0, 256, k0, lane);
    v16bf a1 = load_a_bf16(arow_s1, 256, k0, lane);
#pragma unroll
    for (int j = 0; j < 4; ++j) {
      v16bf b = load_b_bf16(Cw, 256, k0, d0 + 16 * j, lane);  // B[k=n][d] = C_w[d][n]
      acc[0][j] = wmma_bf16(a0, b, acc[0][j]);
      acc[1][j] = wmma_bf16(a1, b, acc[1][j]);
    }
  }
  const unsigned short* arow_x0 = xbf + (size_t)t0 * 512;
  const unsigned short* arow_x1 = arow_x0 + 16 * 512;
#pragma unroll 2
  for (int k0 = 0; k0 < 512; k0 += 32) {
    v16bf a0 = load_a_bf16(arow_x0, 512, k0, lane);
    v16bf a1 = load_a_bf16(arow_x1, 512, k0, lane);
#pragma unroll
    for (int j = 0; j < 4; ++j) {
      v16bf b = load_b_bf16(Dw, 512, k0, d0 + 16 * j, lane);  // B[k=e][d] = D_w[d][e]
      acc[0][j] = wmma_bf16(a0, b, acc[0][j]);
      acc[1][j] = wmma_bf16(a1, b, acc[1][j]);
    }
  }

  if (threadIdx.x < 32) { s_sum[threadIdx.x] = 0.f; s_sq[threadIdx.x] = 0.f; }
  __syncthreads();

  const int hi = (lane < 16) ? 0 : 8;
  const int c  = lane & 15;
  const float* xrow = x + (size_t)t0 * 512;
  // Pass 1: fold residual into acc, accumulate per-token sum / sum-of-squares.
#pragma unroll
  for (int mt = 0; mt < 2; ++mt) {
#pragma unroll
    for (int r = 0; r < 8; ++r) {
      int m = mt * 16 + r + hi;
      const float* xr = xrow + (size_t)m * 512;
      float psum = 0.f, psq = 0.f;
#pragma unroll
      for (int j = 0; j < 4; ++j) {
        float v = acc[mt][j][r] + xr[d0 + 16 * j + c];   // fp32 residual
        acc[mt][j][r] = v;
        psum += v; psq += v * v;
      }
      atomicAdd(&s_sum[m], psum);   // ds_add_f32
      atomicAdd(&s_sq[m],  psq);
    }
  }
  __syncthreads();
  // Pass 2: normalize + affine + store.
#pragma unroll
  for (int mt = 0; mt < 2; ++mt) {
#pragma unroll
    for (int r = 0; r < 8; ++r) {
      int m = mt * 16 + r + hi;
      float mu  = s_sum[m] * (1.f / 512.f);
      float var = s_sq[m] * (1.f / 512.f) - mu * mu;
      float rs  = rsqrtf(var + 1e-5f);
      float* orow = out + (size_t)(t0 + m) * 512;
#pragma unroll
      for (int j = 0; j < 4; ++j) {
        int d = d0 + 16 * j + c;
        orow[d] = (acc[mt][j][r] - mu) * rs * gamma[d] + beta[d];
      }
    }
  }
}

// =================== host launch ===================

extern "C" void kernel_launch(void* const* d_in, const int* in_sizes, int n_in,
                              void* d_out, int out_size, void* d_ws, size_t ws_size,
                              hipStream_t stream) {
  (void)in_sizes; (void)n_in; (void)out_size; (void)ws_size;
  const float* x     = (const float*)d_in[0];   // [64,1024,512]
  const float* A     = (const float*)d_in[1];   // [256,256]
  const float* B_w   = (const float*)d_in[2];   // [256,512]
  const float* C_w   = (const float*)d_in[3];   // [512,256]
  const float* D_w   = (const float*)d_in[4];   // [512,512]
  const float* gamma = (const float*)d_in[5];
  const float* beta  = (const float*)d_in[6];
  float* out = (float*)d_out;

  char* ws = (char*)d_ws;
  unsigned short* Abf  = (unsigned short*)ws;  ws += (size_t)256 * 256 * 2;
  unsigned short* Bwbf = (unsigned short*)ws;  ws += (size_t)256 * 512 * 2;
  unsigned short* Cwbf = (unsigned short*)ws;  ws += (size_t)512 * 256 * 2;
  unsigned short* Dwbf = (unsigned short*)ws;  ws += (size_t)512 * 512 * 2;
  unsigned short* xbf  = (unsigned short*)ws;  ws += (size_t)64 * 1024 * 512 * 2;
  float*          bx   = (float*)ws;           ws += (size_t)1024 * 64 * 256 * 4;
  unsigned short* stbf = (unsigned short*)ws;  ws += (size_t)64 * 1024 * 256 * 2;

  k_f32_to_bf16<<<(256 * 256 + 255) / 256, 256, 0, stream>>>(A,   Abf,  256 * 256);
  k_f32_to_bf16<<<(256 * 512 + 255) / 256, 256, 0, stream>>>(B_w, Bwbf, 256 * 512);
  k_f32_to_bf16<<<(512 * 256 + 255) / 256, 256, 0, stream>>>(C_w, Cwbf, 512 * 256);
  k_f32_to_bf16<<<(512 * 512 + 255) / 256, 256, 0, stream>>>(D_w, Dwbf, 512 * 512);
  k_x_to_bf16<<<(64 * 1024 * 512 / 8 + 255) / 256, 256, 0, stream>>>(x, xbf, 64 * 1024 * 512 / 8);

  k_bx_gemm<<<2048, 256, 0, stream>>>(xbf, Bwbf, bx);

  const int scan_smem = 256 * ALD * 2          // A: 139264 B
                      + 2 * 16 * STLD * 2      // state ping-pong: 17408 B
                      + 2 * 16 * 256 * 4;      // bx double buffer: 32768 B  (~185 KB / 320 KB WGP)
  (void)hipFuncSetAttribute((const void*)k_scan, hipFuncAttributeMaxDynamicSharedMemorySize, scan_smem);
  k_scan<<<4, 256, scan_smem, stream>>>(Abf, bx, stbf);

  k_out_ln<<<2048, 256, 0, stream>>>(stbf, Cwbf, xbf, x, Dwbf, gamma, beta, out);
}
